// CausalSelfAttention_66391604462247
// MI455X (gfx1250) — matrix operand order repaired and checked by reading, not compiled, not run on hardware
//
#include <hip/hip_runtime.h>
#include <hip/hip_bf16.h>

// ---- problem constants (from reference) ----
constexpr int DIMc = 1024;
constexpr int Hq   = 16;
constexpr int HKVc = 4;
constexpr int HDc  = 64;
constexpr int Wwin = 512;
constexpr int Bsz  = 2;
constexpr int Ssz  = 4096;
constexpr int Gc   = Hq / HKVc;     // 4
constexpr int Cblk = Ssz / Wwin;    // 8
constexpr int NTOK = Bsz * Ssz;     // 8192

typedef __bf16 bf16;
typedef __attribute__((ext_vector_type(16))) __bf16 v16bf;
typedef __attribute__((ext_vector_type(8)))  __bf16 v8bf;
typedef __attribute__((ext_vector_type(8)))  float  v8f;

// ---------------------------------------------------------------------------
// WMMA helpers (CDNA5 gfx1250, wave32).  Layouts per ISA 7.12.2:
//  A (16x32 bf16): lane = {m=lane&15, half=lane>>4}; per lane two contiguous
//     8-element runs at K = half*8 and K = 16+half*8.
//  B (32x16 bf16): lane = {n=lane&15, half=lane>>4}; per lane two 8-element runs at
//     K = half*8 and K = 16+half*8, as for A.
//  C/D (16x16 f32): element v of v8f at (m = v + (lane>>4)*8, n = lane&15).
// ---------------------------------------------------------------------------
__device__ inline v8f wmma_bf16(v16bf a, v16bf b, v8f c) {
  return __builtin_amdgcn_wmma_f32_16x16x32_bf16(
      /*neg_a=*/false, a, /*neg_b=*/false, b,
      /*c_mod=*/(short)0, c, /*reuse_a=*/false, /*reuse_b=*/false);
}

__device__ inline v16bf load_a_frag(const bf16* __restrict__ A, int lda,
                                    int mbase, int kbase) {
  const int lane = threadIdx.x & 31;
  const int half = lane >> 4;
  const bf16* p = A + (size_t)(mbase + (lane & 15)) * lda + kbase + half * 8;
  v8bf lo = *(const v8bf*)p;         // K = half*8 .. +7
  v8bf hi = *(const v8bf*)(p + 16);  // K = 16+half*8 .. +7
  v16bf r;
#pragma unroll
  for (int i = 0; i < 8; ++i) { r[i] = lo[i]; r[8 + i] = hi[i]; }
  return r;
}

__device__ inline v16bf load_b_frag(const bf16* __restrict__ Bm, int ldb,
                                    int nbase, int kbase) {
  const int lane = threadIdx.x & 31;
  const bf16* p = Bm + (size_t)(nbase + (lane & 15)) * ldb + kbase +
                  ((lane >> 4) << 3);
  v8bf lo = *(const v8bf*)p;        // K = half*8 .. +7
  v8bf hi = *(const v8bf*)(p + 16); // K = 16+half*8 .. +7
  v16bf r;
#pragma unroll
  for (int i = 0; i < 8; ++i) { r[i] = lo[i]; r[8 + i] = hi[i]; }
  return r;
}

// ---------------------------------------------------------------------------
// fp32 -> bf16 elementwise convert
// ---------------------------------------------------------------------------
__global__ void cvt_f32_bf16(const float* __restrict__ s, bf16* __restrict__ d,
                             int n) {
  int i = blockIdx.x * blockDim.x + threadIdx.x;
  if (i < n) d[i] = (bf16)s[i];
}

// ---------------------------------------------------------------------------
// C[M,N] = A[M,K] * B[N,K]^T  (bf16 in, f32 out).
// Requires M%128==0, N%64==0, K%64==0 and K>=64.
// 256 threads = 8 waves arranged 2(M) x 4(N); each wave owns a 64x64 tile:
// 4 A frags x 4 B frags -> 16 WMMAs per 32-wide K-step.  The K loop is
// unrolled 2x with ping-pong fragment buffers (no register rotation copies):
// loads for one buffer are in flight while WMMAs consume the other.
// ---------------------------------------------------------------------------
__global__ __launch_bounds__(256) void gemm_bf16_nt(
    const bf16* __restrict__ A, const bf16* __restrict__ Bm,
    float* __restrict__ C, int M, int N, int K) {
  const int wave = threadIdx.x >> 5;
  const int lane = threadIdx.x & 31;
  const int mt = blockIdx.y * 128 + (wave & 1) * 64;
  const int nt = blockIdx.x * 256 + (wave >> 1) * 64;
  if (mt >= M || nt >= N) return;

  v8f acc[4][4] = {};
  v16bf a0[4], b0[4], a1[4], b1[4];
#pragma unroll
  for (int i = 0; i < 4; ++i) {
    a0[i] = load_a_frag(A, K, mt + 16 * i, 0);
    b0[i] = load_b_frag(Bm, K, nt + 16 * i, 0);
  }

  int k = 0;
  for (; k + 64 < K; k += 64) {
    __builtin_prefetch(A + (size_t)(mt + (lane & 15)) * K + k + 128, 0, 1);
#pragma unroll
    for (int i = 0; i < 4; ++i) {
      a1[i] = load_a_frag(A, K, mt + 16 * i, k + 32);
      b1[i] = load_b_frag(Bm, K, nt + 16 * i, k + 32);
    }
#pragma unroll
    for (int i = 0; i < 4; ++i)
#pragma unroll
      for (int j = 0; j < 4; ++j)
        acc[i][j] = wmma_bf16(a0[i], b0[j], acc[i][j]);
#pragma unroll
    for (int i = 0; i < 4; ++i) {
      a0[i] = load_a_frag(A, K, mt + 16 * i, k + 64);
      b0[i] = load_b_frag(Bm, K, nt + 16 * i, k + 64);
    }
#pragma unroll
    for (int i = 0; i < 4; ++i)
#pragma unroll
      for (int j = 0; j < 4; ++j)
        acc[i][j] = wmma_bf16(a1[i], b1[j], acc[i][j]);
  }
  // tail: k == K-64, two WMMA groups, one trailing load group
#pragma unroll
  for (int i = 0; i < 4; ++i) {
    a1[i] = load_a_frag(A, K, mt + 16 * i, k + 32);
    b1[i] = load_b_frag(Bm, K, nt + 16 * i, k + 32);
  }
#pragma unroll
  for (int i = 0; i < 4; ++i)
#pragma unroll
    for (int j = 0; j < 4; ++j) acc[i][j] = wmma_bf16(a0[i], b0[j], acc[i][j]);
#pragma unroll
  for (int i = 0; i < 4; ++i)
#pragma unroll
    for (int j = 0; j < 4; ++j) acc[i][j] = wmma_bf16(a1[i], b1[j], acc[i][j]);

  const int mrow = (lane >> 4) * 8;
  const int ncol = lane & 15;
#pragma unroll
  for (int i = 0; i < 4; ++i)
#pragma unroll
    for (int j = 0; j < 4; ++j)
#pragma unroll
      for (int v = 0; v < 8; ++v)
        C[(size_t)(mt + 16 * i + mrow + v) * N + nt + 16 * j + ncol] =
            acc[i][j][v];
}

// ---------------------------------------------------------------------------
// Per-head RMSNorm + RoPE (+ q_gain).  One wave per (token, head)-vector of 64.
// Lane i holds elements (i, i+32) == the RoPE rotation pair.
// q output layout: (B,H,S,HD) bf16; k output layout: (B,HKV,S,HD) bf16.
// ---------------------------------------------------------------------------
__global__ __launch_bounds__(256) void qk_norm_rope(
    const float* __restrict__ qraw, const float* __restrict__ kraw,
    const float* __restrict__ q_gain, bf16* __restrict__ qbf,
    bf16* __restrict__ kbf) {
  const int wid = blockIdx.x * (blockDim.x >> 5) + (threadIdx.x >> 5);
  const int lane = threadIdx.x & 31;
  constexpr int nheads = Hq + HKVc;
  if (wid >= NTOK * nheads) return;
  const int h = wid % nheads;
  const int tok = wid / nheads;
  const int b = tok / Ssz, s = tok % Ssz;

  const float* src;
  bf16* dst;
  float gain = 1.0f;
  if (h < Hq) {
    src = qraw + (size_t)tok * (Hq * HDc) + h * HDc;
    gain = q_gain[h];
    dst = qbf + (((size_t)b * Hq + h) * Ssz + s) * HDc;
  } else {
    const int hk = h - Hq;
    src = kraw + (size_t)tok * (HKVc * HDc) + hk * HDc;
    dst = kbf + (((size_t)b * HKVc + hk) * Ssz + s) * HDc;
  }

  float t1 = src[lane], t2 = src[lane + 32];
  float ss = t1 * t1 + t2 * t2;
#pragma unroll
  for (int o = 16; o; o >>= 1) ss += __shfl_xor(ss, o, 32);
  const float r = rsqrtf(ss * (1.0f / HDc) + 1.1920929e-07f);
  t1 *= r;
  t2 *= r;
  const float freq = (float)s * __powf(10000.0f, -(float)lane * (1.0f / 32.0f));
  float sf, cf;
  __sincosf(freq, &sf, &cf);
  dst[lane]      = (bf16)((t1 * cf + t2 * sf) * gain);
  dst[lane + 32] = (bf16)((-t1 * sf + t2 * cf) * gain);
}

// ---------------------------------------------------------------------------
// vT[b][hk][d][s] = bf16(vraw[b][s][hk][d]) so the P@V B-operand is contiguous
// along the key (contraction) dimension.
// ---------------------------------------------------------------------------
__global__ void v_transpose(const float* __restrict__ vraw,
                            bf16* __restrict__ vT) {
  int i = blockIdx.x * blockDim.x + threadIdx.x;
  if (i >= Bsz * Ssz * HKVc * HDc) return;
  const int d = i % HDc;
  int t = i / HDc;
  const int hk = t % HKVc;
  const int tok = t / HKVc;
  const int b = tok / Ssz, s = tok % Ssz;
  vT[(((size_t)b * HKVc + hk) * HDc + d) * Ssz + s] = (bf16)vraw[i];
}

// ---------------------------------------------------------------------------
// Sliding-window attention: one wave per 16-query tile per (b,h,c).
// Scores for the 2W=1024-key window live in 64 KB LDS (fits 320 KB/WGP).
// Pass1: S = Q*K^T (WMMA) + mask -> LDS.  Pass2: row softmax in LDS.
// Pass3: Y = P*V (WMMA, P repacked bf16 from LDS, V from transposed vT).
// ---------------------------------------------------------------------------
__global__ __launch_bounds__(32) void attn_win(const bf16* __restrict__ qbf,
                                               const bf16* __restrict__ kbf,
                                               const bf16* __restrict__ vT,
                                               float* __restrict__ y) {
  __shared__ float sc[16 * 1024];  // 64 KB
  const int lane = threadIdx.x & 31;
  const int c = blockIdx.x >> 5;
  const int qt = blockIdx.x & 31;
  const int h = blockIdx.y;
  const int b = blockIdx.z;
  const int hk = h / Gc;
  const int qbase = c * Wwin + qt * 16;
  const int kwin = (c - 1) * Wwin;  // window start key index (may be < 0)

  const bf16* Q = qbf + (((size_t)b * Hq + h) * Ssz) * HDc;
  const bf16* Km = kbf + (((size_t)b * HKVc + hk) * Ssz) * HDc;
  const bf16* Vt = vT + (((size_t)b * HKVc + hk) * HDc) * Ssz;

  const v16bf a0 = load_a_frag(Q, HDc, qbase, 0);
  const v16bf a1 = load_a_frag(Q, HDc, qbase, 32);
  const int mrow = (lane >> 4) * 8;
  const int ncol = lane & 15;

  // ---- pass 1: masked scores into LDS ----
  for (int ntile = 0; ntile < 64; ++ntile) {
    const int kb = kwin + ntile * 16;  // multiple of 16
    if (kb < 0) {
#pragma unroll
      for (int v = 0; v < 8; ++v)
        sc[(mrow + v) * 1024 + ntile * 16 + ncol] = -1e30f;
      continue;
    }
    v8f acc = {};
    const v16bf b0 = load_b_frag(Km, HDc, kb, 0);
    const v16bf b1 = load_b_frag(Km, HDc, kb, 32);
    acc = wmma_bf16(a0, b0, acc);
    acc = wmma_bf16(a1, b1, acc);
#pragma unroll
    for (int v = 0; v < 8; ++v) {
      const int qi = qbase + mrow + v;
      const int ki = kb + ncol;
      const bool ok = (ki <= qi) && (ki > qi - Wwin);
      sc[(mrow + v) * 1024 + ntile * 16 + ncol] =
          ok ? acc[v] * 0.125f : -1e30f;
    }
  }
  __syncthreads();

  // ---- pass 2: row softmax (2 lanes per row, 512 cols each) ----
  {
    float* rp = sc + (lane & 15) * 1024 + (lane >> 4) * 512;
    float mx = -1e30f;
    for (int i = 0; i < 512; ++i) mx = fmaxf(mx, rp[i]);
    mx = fmaxf(mx, __shfl_xor(mx, 16, 32));
    float sm = 0.0f;
    for (int i = 0; i < 512; ++i) {
      const float p = __expf(rp[i] - mx);
      rp[i] = p;
      sm += p;
    }
    sm += __shfl_xor(sm, 16, 32);
    const float inv = 1.0f / sm;
    for (int i = 0; i < 512; ++i) rp[i] *= inv;
  }
  __syncthreads();

  // ---- pass 3: Y = P @ V ----
  float* yout = y + (size_t)b * Ssz * (Hq * HDc) + (size_t)h * HDc;
#pragma unroll
  for (int dt = 0; dt < 4; ++dt) {
    v8f acc = {};
    for (int kk = 0; kk < 32; ++kk) {
      const int kb = kwin + kk * 32;  // multiple of 32
      if (kb < 0) continue;           // P == 0 there
      v16bf pa;
      {
        const float* pr = sc + (lane & 15) * 1024 + kk * 32 + (lane >> 4) * 8;
#pragma unroll
        for (int i = 0; i < 8; ++i) {
          pa[i] = (bf16)pr[i];
          pa[8 + i] = (bf16)pr[16 + i];
        }
      }
      v16bf vb;
      {
        const bf16* vp = Vt + (size_t)(dt * 16 + (lane & 15)) * Ssz + kb +
                         ((lane >> 4) << 3);
        v8bf lo = *(const v8bf*)vp;
        v8bf hi = *(const v8bf*)(vp + 16);
#pragma unroll
        for (int i = 0; i < 8; ++i) { vb[i] = lo[i]; vb[8 + i] = hi[i]; }
      }
      acc = wmma_bf16(pa, vb, acc);
    }
#pragma unroll
    for (int v = 0; v < 8; ++v) {
      const int s = qbase + mrow + v;
      yout[(size_t)s * (Hq * HDc) + dt * 16 + ncol] = acc[v];
    }
  }
}

// ---------------------------------------------------------------------------
// Value-direction rejection: one wave per (token, kv-head).
// yg -= dot(yg, vn) * vn for each of G grouped q-heads; emits bf16 for Wp GEMM.
// ---------------------------------------------------------------------------
__global__ __launch_bounds__(256) void v_reject(const float* __restrict__ ybuf,
                                                const float* __restrict__ vraw,
                                                bf16* __restrict__ ybar) {
  const int wid = blockIdx.x * (blockDim.x >> 5) + (threadIdx.x >> 5);
  const int lane = threadIdx.x & 31;
  if (wid >= NTOK * HKVc) return;
  const int hk = wid % HKVc;
  const int tok = wid / HKVc;

  const float* vp = vraw + (size_t)tok * (HKVc * HDc) + hk * HDc;
  float v1 = vp[lane], v2 = vp[lane + 32];
  float nrm = v1 * v1 + v2 * v2;
#pragma unroll
  for (int o = 16; o; o >>= 1) nrm += __shfl_xor(nrm, o, 32);
  const float inv = 1.0f / fmaxf(sqrtf(nrm), 1e-12f);
  v1 *= inv;
  v2 *= inv;

#pragma unroll
  for (int g = 0; g < Gc; ++g) {
    const int h = hk * Gc + g;
    const float* yp = ybuf + ((size_t)tok * Hq + h) * HDc;
    float y1 = yp[lane], y2 = yp[lane + 32];
    float dot = y1 * v1 + y2 * v2;
#pragma unroll
    for (int o = 16; o; o >>= 1) dot += __shfl_xor(dot, o, 32);
    y1 -= dot * v1;
    y2 -= dot * v2;
    bf16* op = ybar + (size_t)tok * DIMc + h * HDc;
    op[lane] = (bf16)y1;
    op[lane + 32] = (bf16)y2;
  }
}

// ---------------------------------------------------------------------------
extern "C" void kernel_launch(void* const* d_in, const int* in_sizes, int n_in,
                              void* d_out, int out_size, void* d_ws,
                              size_t ws_size, hipStream_t stream) {
  const float* x  = (const float*)d_in[0];
  const float* Wq = (const float*)d_in[1];
  const float* Wk = (const float*)d_in[2];
  const float* Wv = (const float*)d_in[3];
  const float* Wp = (const float*)d_in[4];
  const float* qg = (const float*)d_in[5];
  float* out = (float*)d_out;

  // workspace carve-up (256B aligned)
  size_t off = 0;
  auto carve = [&](size_t bytes) {
    void* p = (char*)d_ws + off;
    off = (off + bytes + 255) & ~(size_t)255;
    return p;
  };
  bf16* xbf   = (bf16*)carve((size_t)NTOK * DIMc * 2);
  bf16* wqbf  = (bf16*)carve((size_t)DIMc * DIMc * 2);
  bf16* wkbf  = (bf16*)carve((size_t)HKVc * HDc * DIMc * 2);
  bf16* wvbf  = (bf16*)carve((size_t)HKVc * HDc * DIMc * 2);
  bf16* wpbf  = (bf16*)carve((size_t)DIMc * DIMc * 2);
  float* qraw = (float*)carve((size_t)NTOK * DIMc * 4);
  float* kraw = (float*)carve((size_t)NTOK * HKVc * HDc * 4);
  float* vraw = (float*)carve((size_t)NTOK * HKVc * HDc * 4);
  bf16* qbf   = (bf16*)carve((size_t)NTOK * DIMc * 2);
  bf16* kbf   = (bf16*)carve((size_t)NTOK * HKVc * HDc * 2);
  bf16* vTbf  = (bf16*)carve((size_t)NTOK * HKVc * HDc * 2);
  float* ybuf = (float*)carve((size_t)NTOK * DIMc * 4);
  bf16* ybar  = (bf16*)carve((size_t)NTOK * DIMc * 2);
  (void)ws_size;

  auto cvt = [&](const float* s, bf16* d, int n) {
    cvt_f32_bf16<<<(n + 255) / 256, 256, 0, stream>>>(s, d, n);
  };
  cvt(x, xbf, NTOK * DIMc);
  cvt(Wq, wqbf, DIMc * DIMc);
  cvt(Wk, wkbf, HKVc * HDc * DIMc);
  cvt(Wv, wvbf, HKVc * HDc * DIMc);
  cvt(Wp, wpbf, DIMc * DIMc);

  // QKV projections: out[m][n] = sum_d x[m][d] * W[n][d]
  // block tile = 128(M) x 256(N)
  gemm_bf16_nt<<<dim3(DIMc / 256, NTOK / 128), 256, 0, stream>>>(
      xbf, wqbf, qraw, NTOK, DIMc, DIMc);
  gemm_bf16_nt<<<dim3((HKVc * HDc) / 256, NTOK / 128), 256, 0, stream>>>(
      xbf, wkbf, kraw, NTOK, HKVc * HDc, DIMc);
  gemm_bf16_nt<<<dim3((HKVc * HDc) / 256, NTOK / 128), 256, 0, stream>>>(
      xbf, wvbf, vraw, NTOK, HKVc * HDc, DIMc);

  // norm + rope + gain (q,k), and v transpose for P@V operand
  {
    const int waves = NTOK * (Hq + HKVc);
    qk_norm_rope<<<(waves + 7) / 8, 256, 0, stream>>>(qraw, kraw, qg, qbf, kbf);
    const int n = Bsz * Ssz * HKVc * HDc;
    v_transpose<<<(n + 255) / 256, 256, 0, stream>>>(vraw, vTbf);
  }

  // windowed attention
  attn_win<<<dim3(Cblk * 32, Hq, Bsz), 32, 0, stream>>>(qbf, kbf, vTbf, ybuf);

  // rejection + final projection
  {
    const int waves = NTOK * HKVc;
    v_reject<<<(waves + 7) / 8, 256, 0, stream>>>(ybuf, vraw, ybar);
  }
  gemm_bf16_nt<<<dim3(DIMc / 256, NTOK / 128), 256, 0, stream>>>(
      ybar, wpbf, out, NTOK, DIMc, DIMc);
}
